// MultiHeadAttention_89472758710815
// MI455X (gfx1250) — matrix operand order repaired and checked by reading, not compiled, not run on hardware
//
#include <hip/hip_runtime.h>
#include <hip/hip_bf16.h>

// ---------- types ----------
typedef __attribute__((ext_vector_type(16))) __bf16 v16bf;
typedef __attribute__((ext_vector_type(8)))  __bf16 v8bf;
typedef __attribute__((ext_vector_type(4)))  __bf16 v4bf;
typedef __attribute__((ext_vector_type(8)))  float  v8f;

#define T_SEQ 2048
#define C_DIM 1024
#define NHEAD 16
#define HD    64
#define BATCH 2

static __device__ __forceinline__ v8f wmma_bf16(v16bf a, v16bf b, v8f c) {
  // D = A(16x32 bf16) * B(32x16 bf16) + C(16x16 f32)
  return __builtin_amdgcn_wmma_f32_16x16x32_bf16(false, a, false, b, (short)0, c,
                                                 false, false);
}

static __device__ __forceinline__ v16bf cat8(v8bf lo, v8bf hi) {
  return __builtin_shufflevector(lo, hi, 0,1,2,3,4,5,6,7,8,9,10,11,12,13,14,15);
}

// Low 32 bits of a generic pointer into LDS are the LDS byte offset
// (ISA aperture rule).
static __device__ __forceinline__ unsigned lds_addr32(const void* p) {
  return (unsigned)(uintptr_t)p;
}

// CDNA5 async global->LDS copy, 16 bytes per lane, tracked by ASYNCcnt.
static __device__ __forceinline__ void async_copy_b128(unsigned lds_off,
                                                       const void* gptr) {
  asm volatile("global_load_async_to_lds_b128 %0, %1, off"
               :: "v"(lds_off), "v"((unsigned long long)(uintptr_t)gptr)
               : "memory");
}
static __device__ __forceinline__ void wait_async0() {
  asm volatile("s_wait_asynccnt 0x0" ::: "memory");
}

// ---------- fp32 -> bf16 conversion ----------
__global__ __launch_bounds__(256) void cvt_f32_bf16(const float* __restrict__ in,
                                                    __bf16* __restrict__ out, int n4) {
  int i = blockIdx.x * 256 + threadIdx.x;
  if (i < n4) {
    float4 v = ((const float4*)in)[i];
    v4bf o = { (__bf16)v.x, (__bf16)v.y, (__bf16)v.z, (__bf16)v.w };
    ((v4bf*)out)[i] = o;
  }
}

// ---------- bf16 WMMA GEMM, 128x128 tile, K-step 32, double-buffered ----------
// MODE 0: QKV projection epilogue (scatter Q/K/V per-head, scale Q by 1/sqrt(hd))
// MODE 1: output projection (fp32 out + bias)
template <int MODE>
__global__ __launch_bounds__(256) void gemm_bf16(
    const __bf16* __restrict__ A,   // [M,K] row-major
    const __bf16* __restrict__ Bm,  // [K,N] row-major
    const float*  __restrict__ bias,// [N]
    __bf16* __restrict__ Qd, __bf16* __restrict__ Kd, __bf16* __restrict__ Vt,
    float*  __restrict__ Out,
    int M, int N, int K) {
  __shared__ __bf16 As[2][128][32];   // [m][k]  2 x 8 KB
  __shared__ __bf16 Bs[2][128][32];   // [n][k]  2 x 8 KB (transposed)

  const int tid  = threadIdx.x;
  const int lane = tid & 31;
  const int wave = tid >> 5;
  const int hi   = lane >> 4;   // 0: lanes 0-15, 1: lanes 16-31
  const int lo16 = lane & 15;
  const int m0 = blockIdx.y * 128;
  const int n0 = blockIdx.x * 128;
  const int wm = (wave >> 1) * 32;   // 4 wave-rows of 32
  const int wn = (wave & 1) * 64;    // 2 wave-cols of 64

  v8f acc[2][4];
  const v8f zero8 = {0.f,0.f,0.f,0.f,0.f,0.f,0.f,0.f};
#pragma unroll
  for (int mi = 0; mi < 2; ++mi)
#pragma unroll
    for (int ni = 0; ni < 4; ++ni) acc[mi][ni] = zero8;

  // A tile: straight per-lane 16B async copies into LDS
  auto stageA = [&](int buf, int kt) {
#pragma unroll
    for (int it = 0; it < 2; ++it) {
      int lin = tid + 256 * it;          // 0..511
      int r  = lin >> 2;                 // 4 chunks of 8 per row
      int c8 = lin & 3;
      async_copy_b128(lds_addr32(&As[buf][r][c8 * 8]),
                      A + (size_t)(m0 + r) * K + kt + c8 * 8);
    }
  };
  // B tile needs a transpose -> global loads to VGPR, scatter-store later
  v8bf breg[2];
  auto loadB = [&](int kt) {
#pragma unroll
    for (int it = 0; it < 2; ++it) {
      int lin = tid + 256 * it;
      int kk = lin >> 4, c8 = lin & 15;
      breg[it] = *(const v8bf*)(Bm + (size_t)(kt + kk) * N + n0 + c8 * 8);
    }
  };
  auto storeB = [&](int buf) {
#pragma unroll
    for (int it = 0; it < 2; ++it) {
      int lin = tid + 256 * it;
      int kk = lin >> 4, c8 = lin & 15;
#pragma unroll
      for (int j = 0; j < 8; ++j) Bs[buf][c8 * 8 + j][kk] = breg[it][j];
    }
  };

  // prologue: stage tile 0
  stageA(0, 0);
  loadB(0);
  storeB(0);
  wait_async0();
  __syncthreads();

  int buf = 0;
  for (int kt = 0; kt < K; kt += 32, buf ^= 1) {
    const int ktn = kt + 32;
    if (ktn < K) {            // start next tile's copies before computing
      stageA(buf ^ 1, ktn);
      loadB(ktn);
    }

    // A fragments: lane holds row M=lane%16; K = kb .. kb+15 contiguous
    v16bf afrag[2];
#pragma unroll
    for (int mi = 0; mi < 2; ++mi) {
      int r  = wm + mi * 16 + lo16;
      int kb = hi * 16;
      afrag[mi] = *(const v16bf*)&As[buf][r][kb];
    }
    // B fragments: lane holds col N=lane%16; K = hi*16 .. hi*16+15 contiguous
#pragma unroll
    for (int ni = 0; ni < 4; ++ni) {
      int cidx = wn + ni * 16 + lo16;
      v16bf bfrag = *(const v16bf*)&Bs[buf][cidx][hi * 16];
#pragma unroll
      for (int mi = 0; mi < 2; ++mi)
        acc[mi][ni] = wmma_bf16(afrag[mi], bfrag, acc[mi][ni]);
    }

    if (ktn < K) storeB(buf ^ 1);
    wait_async0();
    __syncthreads();
  }

  // epilogue: C layout = VGPR i -> row (i + hi*8), col = lane%16
#pragma unroll
  for (int mi = 0; mi < 2; ++mi) {
#pragma unroll
    for (int ni = 0; ni < 4; ++ni) {
      int n = n0 + wn + ni * 16 + lo16;
      float bv = bias[n];
#pragma unroll
      for (int i = 0; i < 8; ++i) {
        int m = m0 + wm + mi * 16 + hi * 8 + i;
        float val = acc[mi][ni][i] + bv;
        if (MODE == 0) {
          // column n of the 3C-wide qkv: head h, chunk {q,k,v}, dim d
          int h = n / 192;
          int rem = n - h * 192;
          int which = rem >> 6;
          int d = rem & 63;
          int b = m >> 11;          // m = b*T + t
          int t = m & (T_SEQ - 1);
          size_t roff = ((size_t)(b * NHEAD + h) * T_SEQ + t) * HD + d;
          if (which == 0)
            Qd[roff] = (__bf16)(val * 0.125f);   // fold 1/sqrt(64) into Q
          else if (which == 1)
            Kd[roff] = (__bf16)val;
          else
            Vt[((size_t)(b * NHEAD + h) * HD + d) * T_SEQ + t] = (__bf16)val; // V^T
        } else {
          Out[(size_t)m * N + n] = val;
        }
      }
    }
  }
}

// ---------- flash attention: 1 block per (bh, 128 query rows) ----------
__global__ __launch_bounds__(256) void attn_kernel(
    const __bf16* __restrict__ Qd,  // [BH][T][HD]  (pre-scaled)
    const __bf16* __restrict__ Kd,  // [BH][T][HD]
    const __bf16* __restrict__ Vt,  // [BH][HD][T]
    __bf16* __restrict__ AO) {      // [B*T][C]
  __shared__ __bf16 Ks[2][64][64];     // [key][d]   2 x 8 KB
  __shared__ __bf16 Vs[2][64][64];     // [d][key]   2 x 8 KB
  __shared__ __bf16 Ps[8][16][64];     // per-wave P transpose buffer, 16 KB

  const int tid  = threadIdx.x;
  const int lane = tid & 31;
  const int wave = tid >> 5;
  const int hi   = lane >> 4;
  const int lo16 = lane & 15;
  const int q0 = blockIdx.x * 128;
  const int bh = blockIdx.y;
  const int qw = q0 + wave * 16;     // this wave's 16 query rows

  const __bf16* Qb = Qd + (size_t)bh * T_SEQ * HD;
  const __bf16* Kb = Kd + (size_t)bh * T_SEQ * HD;
  const __bf16* Vb = Vt + (size_t)bh * HD * T_SEQ;

  // async-stage one 64-key K block (contiguous 8 KB) + V^T block
  auto stageKV = [&](int buf, int j) {
#pragma unroll
    for (int it = 0; it < 2; ++it) {
      int lin = tid + 256 * it;  // 0..511, 16B chunks
      async_copy_b128(lds_addr32(((__bf16*)&Ks[buf][0][0]) + lin * 8),
                      Kb + (size_t)j * 64 * HD + lin * 8);
      int d = lin >> 3, c = lin & 7;
      async_copy_b128(lds_addr32(&Vs[buf][d][c * 8]),
                      Vb + (size_t)d * T_SEQ + j * 64 + c * 8);
    }
  };

  // Q fragments stay in registers across the whole key loop
  v16bf aQ[2];
  {
    int r  = qw + lo16;
    int kb = hi * 16;
#pragma unroll
    for (int c = 0; c < 2; ++c)
      aQ[c] = *(const v16bf*)(Qb + (size_t)r * HD + c * 32 + kb);
  }

  float mrow[8], lrow[8];
  v8f Oacc[4];
  const v8f zero8 = {0.f,0.f,0.f,0.f,0.f,0.f,0.f,0.f};
#pragma unroll
  for (int i = 0; i < 8; ++i) { mrow[i] = -INFINITY; lrow[i] = 0.f; }
#pragma unroll
  for (int nh = 0; nh < 4; ++nh) Oacc[nh] = zero8;

  const int jend = (q0 + 127) >> 6;   // causal upper block (inclusive)

  stageKV(0, 0);
  wait_async0();
  __syncthreads();

  for (int j = 0; j <= jend; ++j) {
    const int cur = j & 1;
    if (j + 1 <= jend) stageKV(cur ^ 1, j + 1);  // overlap copy with compute

    if (j * 64 <= qw + 15) {           // wave-uniform causal skip
      // S = Q K^T  (16 x 64 fp32)
      v8f S[4];
#pragma unroll
      for (int ni = 0; ni < 4; ++ni) S[ni] = zero8;
#pragma unroll
      for (int ni = 0; ni < 4; ++ni) {
        int key = ni * 16 + lo16;
#pragma unroll
        for (int c = 0; c < 2; ++c) {
          v16bf bfrag = *(const v16bf*)&Ks[cur][key][c * 32 + hi * 16];
          S[ni] = wmma_bf16(aQ[c], bfrag, S[ni]);
        }
      }
      // causal mask + running row max
      float newm[8];
#pragma unroll
      for (int i = 0; i < 8; ++i) newm[i] = mrow[i];
#pragma unroll
      for (int ni = 0; ni < 4; ++ni) {
        int key = j * 64 + ni * 16 + lo16;
#pragma unroll
        for (int i = 0; i < 8; ++i) {
          int qr = qw + hi * 8 + i;
          float v = S[ni][i];
          if (key > qr) v = -INFINITY;
          S[ni][i] = v;
          newm[i] = fmaxf(newm[i], v);
        }
      }
#pragma unroll
      for (int off = 1; off < 16; off <<= 1)
#pragma unroll
        for (int i = 0; i < 8; ++i)
          newm[i] = fmaxf(newm[i], __shfl_xor(newm[i], off, 32));

      // P = exp(S - m_new), row sums, store P (bf16) for layout transpose
      float psum[8] = {0.f,0.f,0.f,0.f,0.f,0.f,0.f,0.f};
#pragma unroll
      for (int ni = 0; ni < 4; ++ni)
#pragma unroll
        for (int i = 0; i < 8; ++i) {
          float p = __expf(S[ni][i] - newm[i]);
          S[ni][i] = p;
          psum[i] += p;
          Ps[wave][hi * 8 + i][ni * 16 + lo16] = (__bf16)p;
        }
#pragma unroll
      for (int off = 1; off < 16; off <<= 1)
#pragma unroll
        for (int i = 0; i < 8; ++i)
          psum[i] += __shfl_xor(psum[i], off, 32);

      // online rescale of O and l
      float fr[8];
#pragma unroll
      for (int i = 0; i < 8; ++i) {
        fr[i] = __expf(mrow[i] - newm[i]);
        lrow[i] = lrow[i] * fr[i] + psum[i];
        mrow[i] = newm[i];
      }
#pragma unroll
      for (int nh = 0; nh < 4; ++nh)
#pragma unroll
        for (int i = 0; i < 8; ++i) Oacc[nh][i] *= fr[i];

      // O += P V   (contract over 64 keys: 2 chunks of 32)
#pragma unroll
      for (int c2 = 0; c2 < 2; ++c2) {
        int kb = hi * 16;
        v16bf aP = *(const v16bf*)&Ps[wave][lo16][c2 * 32 + kb];
#pragma unroll
        for (int nh = 0; nh < 4; ++nh) {
          int d = nh * 16 + lo16;
          v16bf bV = *(const v16bf*)&Vs[cur][d][c2 * 32 + hi * 16];
          Oacc[nh] = wmma_bf16(aP, bV, Oacc[nh]);
        }
      }
    }

    wait_async0();     // next block's K/V landed in LDS
    __syncthreads();   // everyone done with cur + staged nxt
  }

  // write O / l into (B,T,H,hd) == (B*T, C) bf16 for the projection GEMM
  const int b = bh >> 4, h = bh & 15;
#pragma unroll
  for (int nh = 0; nh < 4; ++nh) {
    int d = nh * 16 + lo16;
#pragma unroll
    for (int i = 0; i < 8; ++i) {
      int t = qw + hi * 8 + i;
      float val = Oacc[nh][i] / lrow[i];
      AO[((size_t)(b * T_SEQ + t)) * C_DIM + h * HD + d] = (__bf16)val;
    }
  }
}

// ---------- host launch ----------
extern "C" void kernel_launch(void* const* d_in, const int* in_sizes, int n_in,
                              void* d_out, int out_size, void* d_ws, size_t ws_size,
                              hipStream_t stream) {
  const float* x     = (const float*)d_in[0];  // [B,T,C]
  const float* Wqkv  = (const float*)d_in[1];  // [C,3C]
  const float* bqkv  = (const float*)d_in[2];  // [3C]
  const float* Wproj = (const float*)d_in[3];  // [C,C]
  const float* bproj = (const float*)d_in[4];  // [C]
  float* out = (float*)d_out;                  // [B,T,C] fp32

  const size_t nx    = (size_t)BATCH * T_SEQ * C_DIM;       // 4 Mi
  const size_t nWq   = (size_t)C_DIM * 3 * C_DIM;           // 3 Mi
  const size_t nWp   = (size_t)C_DIM * C_DIM;               // 1 Mi
  const size_t nHead = (size_t)BATCH * NHEAD * T_SEQ * HD;  // 4 Mi

  char* ws = (char*)d_ws;
  size_t off = 0;
  auto carve = [&](size_t bytes) -> void* {
    void* p = ws + off;
    off += (bytes + 255) & ~(size_t)255;
    return p;
  };
  __bf16* xb  = (__bf16*)carve(nx * 2);
  __bf16* Wqb = (__bf16*)carve(nWq * 2);
  __bf16* Wpb = (__bf16*)carve(nWp * 2);
  __bf16* Qd  = (__bf16*)carve(nHead * 2);
  __bf16* Kd  = (__bf16*)carve(nHead * 2);
  __bf16* Vt  = (__bf16*)carve(nHead * 2);
  __bf16* AO  = (__bf16*)carve(nx * 2);

  // 1) fp32 -> bf16 conversions
  cvt_f32_bf16<<<(int)(nx / 4 / 256), 256, 0, stream>>>(x, xb, (int)(nx / 4));
  cvt_f32_bf16<<<(int)(nWq / 4 / 256), 256, 0, stream>>>(Wqkv, Wqb, (int)(nWq / 4));
  cvt_f32_bf16<<<(int)(nWp / 4 / 256), 256, 0, stream>>>(Wproj, Wpb, (int)(nWp / 4));

  // 2) QKV projection + per-head scatter (Q scaled, V transposed)
  gemm_bf16<0><<<dim3(3 * C_DIM / 128, BATCH * T_SEQ / 128), 256, 0, stream>>>(
      xb, Wqb, bqkv, Qd, Kd, Vt, nullptr,
      BATCH * T_SEQ, 3 * C_DIM, C_DIM);

  // 3) causal flash attention
  attn_kernel<<<dim3(T_SEQ / 128, BATCH * NHEAD), 256, 0, stream>>>(Qd, Kd, Vt, AO);

  // 4) output projection -> fp32 d_out
  gemm_bf16<1><<<dim3(C_DIM / 128, BATCH * T_SEQ / 128), 256, 0, stream>>>(
      AO, Wpb, bproj, nullptr, nullptr, nullptr, out,
      BATCH * T_SEQ, C_DIM, C_DIM);
}
